// MoELoRALinear_68092411511613
// MI455X (gfx1250) — compile-verified, matrix operand order checked by reading, not run on hardware
//
#include <hip/hip_runtime.h>

typedef float    f32x4 __attribute__((ext_vector_type(4)));
typedef float    v8f   __attribute__((ext_vector_type(8)));
typedef _Float16 half8 __attribute__((ext_vector_type(8)));
typedef _Float16 v16h  __attribute__((ext_vector_type(16)));
typedef int      i32x4 __attribute__((vector_size(16)));   // matches builtin param type

#define IN_F   2048
#define OUT_F  2048
#define NTOK   8192          // B*S = 4*2048
#define KLORA  32            // E*R
#define LSCALE 2.0f          // alpha/r = 16/8

#define BM 128
#define BN 128
#define BK 32
#define LDSH 40              // halfs per LDS row: 32 + 8 pad (80B stride, conflict-free)

// ---------------- CDNA5 async global->LDS path (guarded) -------------------
#if defined(__has_builtin)
#  if __has_builtin(__builtin_amdgcn_global_load_async_to_lds_b128)
#    define HAVE_ASYNC_LDS 1
#  endif
#  if __has_builtin(__builtin_amdgcn_s_wait_asynccnt)
#    define HAVE_WAIT_ASYNC 1
#  endif
#endif
#ifndef HAVE_ASYNC_LDS
#  define HAVE_ASYNC_LDS 0
#endif
#ifndef HAVE_WAIT_ASYNC
#  define HAVE_WAIT_ASYNC 0
#endif

typedef __attribute__((address_space(1))) i32x4* gvec_ptr;
typedef __attribute__((address_space(3))) i32x4* lvec_ptr;

template <int N>
__device__ __forceinline__ void wait_async()
{
#if HAVE_WAIT_ASYNC
    __builtin_amdgcn_s_wait_asynccnt(N);
#else
    asm volatile("s_wait_asynccnt %0" ::"i"(N) : "memory");
#endif
}

// copy 8 halfs (16B) global -> LDS, either async (ASYNCcnt) or via VGPRs
__device__ __forceinline__ void copy16B(_Float16* l, const _Float16* g)
{
#if HAVE_ASYNC_LDS
    __builtin_amdgcn_global_load_async_to_lds_b128(
        (gvec_ptr)(uintptr_t)g, (lvec_ptr)l, 0, 0);
#else
    *(half8*)l = *(const half8*)g;
#endif
}

// ---------------------------------------------------------------------------
// split one f32 vector of 8 into hi/lo f16 halves
// ---------------------------------------------------------------------------
__device__ __forceinline__ void split8(const float* src, half8& hi, half8& lo)
{
    f32x4 v0 = *(const f32x4*)(src);
    f32x4 v1 = *(const f32x4*)(src + 4);
#pragma unroll
    for (int c = 0; c < 4; ++c) {
        _Float16 h = (_Float16)v0[c];
        hi[c] = h; lo[c] = (_Float16)(v0[c] - (float)h);
    }
#pragma unroll
    for (int c = 0; c < 4; ++c) {
        _Float16 h = (_Float16)v1[c];
        hi[4 + c] = h; lo[4 + c] = (_Float16)(v1[c] - (float)h);
    }
}

// ---------------------------------------------------------------------------
// Kernel 0: f32 -> (hi,lo) f16 pre-conversion, 8 elements per thread
// ---------------------------------------------------------------------------
__launch_bounds__(256)
__global__ void cvt_kernel(const float* __restrict__ src,
                           _Float16* __restrict__ hi, _Float16* __restrict__ lo)
{
    size_t i = ((size_t)blockIdx.x * 256 + threadIdx.x) * 8;
    half8 h, l;
    split8(src + i, h, l);
    *(half8*)(hi + i) = h;
    *(half8*)(lo + i) = l;
}

// ---------------------------------------------------------------------------
// Kernel 1: router softmax gates + rank-8 expert projections ->
//           P[m][e*8+r] = SCALE * gate[e] * (x[m] . A[e][r])  (as hi/lo f16)
// ---------------------------------------------------------------------------
__launch_bounds__(256)
__global__ void router_proj_kernel(const float* __restrict__ X,
                                   const float* __restrict__ RW,   // (4, 2048)
                                   const float* __restrict__ A,    // (4, 8, 2048)
                                   _Float16* __restrict__ Phi,
                                   _Float16* __restrict__ Plo)
{
    const int lane = threadIdx.x & 31;
    const int wave = threadIdx.x >> 5;
    const int m    = blockIdx.x * 8 + wave;

    __shared__ float red[8][36];

    const f32x4* x4 = (const f32x4*)(X + (size_t)m * IN_F);
    f32x4 xr[16];
#pragma unroll
    for (int j = 0; j < 16; ++j) xr[j] = x4[j * 32 + lane];

    for (int rrow = 0; rrow < 36; ++rrow) {
        const float* wp = (rrow < 4) ? (RW + (size_t)rrow * IN_F)
                                     : (A  + (size_t)(rrow - 4) * IN_F);
        const f32x4* w4 = (const f32x4*)wp;
        float acc = 0.0f;
#pragma unroll
        for (int j = 0; j < 16; ++j) {
            f32x4 wv = w4[j * 32 + lane];
            acc = fmaf(xr[j][0], wv[0], acc);
            acc = fmaf(xr[j][1], wv[1], acc);
            acc = fmaf(xr[j][2], wv[2], acc);
            acc = fmaf(xr[j][3], wv[3], acc);
        }
#pragma unroll
        for (int off = 16; off > 0; off >>= 1)
            acc += __shfl_xor(acc, off, 32);
        if (lane == 0) red[wave][rrow] = acc;
    }
    __syncthreads();

    float l0 = red[wave][0], l1 = red[wave][1], l2 = red[wave][2], l3 = red[wave][3];
    float mx = fmaxf(fmaxf(l0, l1), fmaxf(l2, l3));
    float e0 = __expf(l0 - mx), e1 = __expf(l1 - mx), e2 = __expf(l2 - mx), e3 = __expf(l3 - mx);
    float inv = 1.0f / (e0 + e1 + e2 + e3);
    float g[4] = {e0 * inv, e1 * inv, e2 * inv, e3 * inv};

    const int e = lane >> 3;
    float v = LSCALE * g[e] * red[wave][4 + lane];
    _Float16 h = (_Float16)v;
    Phi[(size_t)m * KLORA + lane] = h;
    Plo[(size_t)m * KLORA + lane] = (_Float16)(v - (float)h);
}

// ---------------------------------------------------------------------------
// Kernel 2: BcatT[n][e*8+r] = Bm[e][n][r], emitted as hi/lo f16
// ---------------------------------------------------------------------------
__launch_bounds__(256)
__global__ void bcat_kernel(const float* __restrict__ Bm,
                            _Float16* __restrict__ BThi, _Float16* __restrict__ BTlo)
{
    int i = blockIdx.x * 256 + threadIdx.x;        // over 2048*32
    int n = i >> 5;
    int k = i & 31;
    int e = k >> 3;
    int r = k & 7;
    float v = Bm[((size_t)e * OUT_F + n) * 8 + r];
    _Float16 h = (_Float16)v;
    BThi[i] = h;
    BTlo[i] = (_Float16)(v - (float)h);
}

// ---------------------------------------------------------------------------
// Fragment loader from padded LDS tile: lanes 0-15 -> rows, K {kb..kb+7} in
// v0..3 and {16+kb..} in v4..7, kb = (lane>=16)*8. Two ds_load_b128.
// ---------------------------------------------------------------------------
__device__ __forceinline__ v16h load_frag(const _Float16* tile, int rowbase, int lane)
{
    const int r  = rowbase + (lane & 15);
    const int kb = (lane >> 4) * 8;
    const _Float16* p = tile + r * LDSH + kb;
    v16h f;
    ((half8*)&f)[0] = *(const half8*)(p);
    ((half8*)&f)[1] = *(const half8*)(p + 16);
    return f;
}

// ---------------------------------------------------------------------------
// Kernel 3: split-f16 WMMA GEMM, double-buffered async global->LDS staging,
// fused LoRA K=32 chunk + bias.
// out[m][n] = sum_k x[m][k]W[n][k] + bias[n] + sum_kk P[m][kk]BcatT[n][kk]
// ---------------------------------------------------------------------------
__launch_bounds__(256, 1)
__global__ void gemm_kernel(const _Float16* __restrict__ Xhi, const _Float16* __restrict__ Xlo,
                            const _Float16* __restrict__ Whi, const _Float16* __restrict__ Wlo,
                            const _Float16* __restrict__ Phi, const _Float16* __restrict__ Plo,
                            const _Float16* __restrict__ BThi, const _Float16* __restrict__ BTlo,
                            const float* __restrict__ bias, float* __restrict__ out)
{
    // [buf][Ahi,Alo,Bhi,Blo][128 rows * LDSH]
    __shared__ __align__(16) _Float16 smem[2][4][BM * LDSH];

    const int tid  = threadIdx.x;
    const int lane = tid & 31;
    const int wave = tid >> 5;
    const int wm   = wave >> 2;          // 0..1 -> 64 rows of M
    const int wn   = wave & 3;           // 0..3 -> 32 cols of N
    const int bm0  = blockIdx.y * BM;
    const int bn0  = blockIdx.x * BN;

    const int NCHUNK = IN_F / BK;        // 64 main chunks; chunk 64 = LoRA

    v8f acc[4][2];
#pragma unroll
    for (int i = 0; i < 4; ++i)
#pragma unroll
        for (int j = 0; j < 2; ++j) acc[i][j] = (v8f)0.0f;

    const int lrow = tid >> 1;           // 0..127
    const int lcol = (tid & 1) * 16;     // 0 or 16

    auto issue_chunk = [&](int kt, int buf) {
        const _Float16 *sAh, *sAl, *sBh, *sBl;
        int lda, ldb, k0;
        if (kt < NCHUNK) {
            sAh = Xhi + (size_t)bm0 * IN_F;  sAl = Xlo + (size_t)bm0 * IN_F;  lda = IN_F;
            sBh = Whi + (size_t)bn0 * IN_F;  sBl = Wlo + (size_t)bn0 * IN_F;  ldb = IN_F;
            k0 = kt * BK;
        } else {
            sAh = Phi + (size_t)bm0 * KLORA;  sAl = Plo + (size_t)bm0 * KLORA;  lda = KLORA;
            sBh = BThi + (size_t)bn0 * KLORA; sBl = BTlo + (size_t)bn0 * KLORA; ldb = KLORA;
            k0 = 0;
        }
        const size_t goffA = (size_t)lrow * lda + k0 + lcol;
        const size_t goffB = (size_t)lrow * ldb + k0 + lcol;
        const int    loff  = lrow * LDSH + lcol;

        _Float16* l0 = &smem[buf][0][loff];
        copy16B(l0,     sAh + goffA);
        copy16B(l0 + 8, sAh + goffA + 8);
        _Float16* l1 = &smem[buf][1][loff];
        copy16B(l1,     sAl + goffA);
        copy16B(l1 + 8, sAl + goffA + 8);
        _Float16* l2 = &smem[buf][2][loff];
        copy16B(l2,     sBh + goffB);
        copy16B(l2 + 8, sBh + goffB + 8);
        _Float16* l3 = &smem[buf][3][loff];
        copy16B(l3,     sBl + goffB);
        copy16B(l3 + 8, sBl + goffB + 8);
    };

    issue_chunk(0, 0);

    for (int kt = 0; kt <= NCHUNK; ++kt) {
        const int buf = kt & 1;
        if (kt < NCHUNK) {
            issue_chunk(kt + 1, buf ^ 1);   // prefetch next chunk into other buffer
            wait_async<8>();                // our 8 prefetch copies may stay in flight
        } else {
            wait_async<0>();
        }
        __syncthreads();                    // all waves' copies for chunk kt landed

        const _Float16* sAhi = smem[buf][0];
        const _Float16* sAlo = smem[buf][1];
        const _Float16* sBhi = smem[buf][2];
        const _Float16* sBlo = smem[buf][3];

        v16h bhi0 = load_frag(sBhi, wn * 32 + 0,  lane);
        v16h blo0 = load_frag(sBlo, wn * 32 + 0,  lane);
        v16h bhi1 = load_frag(sBhi, wn * 32 + 16, lane);
        v16h blo1 = load_frag(sBlo, wn * 32 + 16, lane);

#pragma unroll
        for (int mi = 0; mi < 4; ++mi) {
            v16h ahi = load_frag(sAhi, wm * 64 + mi * 16, lane);
            v16h alo = load_frag(sAlo, wm * 64 + mi * 16, lane);

            acc[mi][0] = __builtin_amdgcn_wmma_f32_16x16x32_f16(false, ahi, false, bhi0, (short)0, acc[mi][0], false, false);
            acc[mi][0] = __builtin_amdgcn_wmma_f32_16x16x32_f16(false, ahi, false, blo0, (short)0, acc[mi][0], false, false);
            acc[mi][0] = __builtin_amdgcn_wmma_f32_16x16x32_f16(false, alo, false, bhi0, (short)0, acc[mi][0], false, false);

            acc[mi][1] = __builtin_amdgcn_wmma_f32_16x16x32_f16(false, ahi, false, bhi1, (short)0, acc[mi][1], false, false);
            acc[mi][1] = __builtin_amdgcn_wmma_f32_16x16x32_f16(false, ahi, false, blo1, (short)0, acc[mi][1], false, false);
            acc[mi][1] = __builtin_amdgcn_wmma_f32_16x16x32_f16(false, alo, false, bhi1, (short)0, acc[mi][1], false, false);
        }
        __syncthreads();                    // chunk kt's buffer free for reuse
    }

    // epilogue: C layout VGPR r -> M = r + 8*(lane>=16), N = lane&15
#pragma unroll
    for (int mi = 0; mi < 4; ++mi) {
#pragma unroll
        for (int ni = 0; ni < 2; ++ni) {
            const int n  = bn0 + wn * 32 + ni * 16 + (lane & 15);
            const float bv = bias[n];
            const int mb = bm0 + wm * 64 + mi * 16 + ((lane >> 4) << 3);
#pragma unroll
            for (int r = 0; r < 8; ++r) {
                out[(size_t)(mb + r) * OUT_F + n] = acc[mi][ni][r] + bv;
            }
        }
    }
}

// ---------------------------------------------------------------------------
// launch
// ---------------------------------------------------------------------------
extern "C" void kernel_launch(void* const* d_in, const int* in_sizes, int n_in,
                              void* d_out, int out_size, void* d_ws, size_t ws_size,
                              hipStream_t stream)
{
    const float* x        = (const float*)d_in[0];   // (4,2048,2048)
    const float* base_w   = (const float*)d_in[1];   // (2048,2048)
    const float* base_b   = (const float*)d_in[2];   // (2048,)
    const float* A        = (const float*)d_in[3];   // (4,8,2048)
    const float* Bm       = (const float*)d_in[4];   // (4,2048,8)
    const float* router_w = (const float*)d_in[5];   // (4,2048)
    float* out = (float*)d_out;

    // workspace layout (halfs): Xhi|Xlo|Whi|Wlo|Phi|Plo|BThi|BTlo  (~85 MB)
    _Float16* Xhi  = (_Float16*)d_ws;
    _Float16* Xlo  = Xhi + (size_t)NTOK * IN_F;
    _Float16* Whi  = Xlo + (size_t)NTOK * IN_F;
    _Float16* Wlo  = Whi + (size_t)OUT_F * IN_F;
    _Float16* Phi  = Wlo + (size_t)OUT_F * IN_F;
    _Float16* Plo  = Phi + (size_t)NTOK * KLORA;
    _Float16* BThi = Plo + (size_t)NTOK * KLORA;
    _Float16* BTlo = BThi + (size_t)OUT_F * KLORA;

    cvt_kernel<<<(NTOK * (IN_F / 8)) / 256, 256, 0, stream>>>(x, Xhi, Xlo);
    cvt_kernel<<<(OUT_F * (IN_F / 8)) / 256, 256, 0, stream>>>(base_w, Whi, Wlo);
    router_proj_kernel<<<NTOK / 8, 256, 0, stream>>>(x, router_w, A, Phi, Plo);
    bcat_kernel<<<(OUT_F * KLORA) / 256, 256, 0, stream>>>(Bm, BThi, BTlo);
    gemm_kernel<<<dim3(OUT_F / BN, NTOK / BM), 256, 0, stream>>>(
        Xhi, Xlo, Whi, Wlo, Phi, Plo, BThi, BTlo, base_b, out);
}
